// att_with_spp_57990648430700
// MI455X (gfx1250) — compile-verified, matrix operand order checked by reading, not run on hardware
//
#include <hip/hip_runtime.h>
#include <cstdint>
#include <cstddef>

// Problem constants
#define BB 256
#define CC 512
#define NN 49            // 7*7 attention positions
#define MM (BB * NN)     // 12544 rows of the big GEMM
#define MT 2             // 16-row M-subtiles per block in the fused GEMM

typedef __attribute__((ext_vector_type(16))) __bf16 v16bf;
typedef __attribute__((ext_vector_type(8)))  float  v8f;

struct U16x2 { uint4 lo, hi; };   // 32 bytes == 16 bf16

// Pin memory + WMMA ops across this point; VALU/SALU may still cross.
#if defined(__has_builtin)
#if __has_builtin(__builtin_amdgcn_sched_barrier)
#define SCHED_FENCE() __builtin_amdgcn_sched_barrier(0x0006)
#endif
#endif
#ifndef SCHED_FENCE
#define SCHED_FENCE() ((void)0)
#endif

__device__ __forceinline__ uint16_t f32_to_bf16(float f) {
    uint32_t u = __builtin_bit_cast(uint32_t, f);
    u += 0x7fffu + ((u >> 16) & 1u);   // round-to-nearest-even
    return (uint16_t)(u >> 16);
}

// Load one 16-element bf16 fragment: 8 elems at p, 8 elems at p+16
// (matches CDNA5 16-bit A/B VGPR layout: K groups {kh*8..kh*8+7} and {16+kh*8..})
__device__ __forceinline__ v16bf load_frag(const uint16_t* p) {
    U16x2 t;
    t.lo = *(const uint4*)(p);
    t.hi = *(const uint4*)(p + 16);
    return __builtin_bit_cast(v16bf, t);
}

// ---------------------------------------------------------------------------
// K1: 2x2 block-mean pool. att_v[B,C,14,14] -> a[m=b*49+n][k=c] (f32 + bf16).
// grid: B*8 blocks (64 channels each), block 256.
// ---------------------------------------------------------------------------
__global__ void __launch_bounds__(256)
pool_kernel(const float* __restrict__ att_v,
            float* __restrict__ a_f32, uint16_t* __restrict__ a_bf16) {
    __shared__ float lds[64 * 196];
    const int blk = blockIdx.x;
    const int b  = blk >> 3;
    const int c0 = (blk & 7) * 64;
    const float* src = att_v + ((size_t)b * CC + c0) * 196;
    for (int i = threadIdx.x; i < 64 * 196; i += 256) lds[i] = src[i];
    __syncthreads();
    for (int j = threadIdx.x; j < NN * 64; j += 256) {
        const int c = j & 63;
        const int n = j >> 6;
        const int s1 = n / 7, s2 = n % 7;
        const float* p = &lds[c * 196 + (2 * s1) * 14 + 2 * s2];
        const float m = 0.25f * (p[0] + p[1] + p[14] + p[15]);
        const size_t idx = ((size_t)(b * NN + n)) * CC + c0 + c;
        a_f32[idx]  = m;
        a_bf16[idx] = f32_to_bf16(m);
    }
}

// ---------------------------------------------------------------------------
// K2: f32 -> bf16 convert
// ---------------------------------------------------------------------------
__global__ void __launch_bounds__(256)
cvt_kernel(const float* __restrict__ in, uint16_t* __restrict__ out, int n) {
    const int i = blockIdx.x * 256 + threadIdx.x;
    if (i < n) out[i] = f32_to_bf16(in[i]);
}

// ---------------------------------------------------------------------------
// K3: out[M x 512] = A[M x 512](bf16) @ W[512 x 512](bf16, row-major, used ^T)
//     + bias.  grid = M/16 blocks, block = 256 (8 waves, 4 N-tiles per wave).
//     B fragments software-pipelined one k-step ahead; sched_barrier keeps
//     the prefetch loads ahead of the WMMA block.
// ---------------------------------------------------------------------------
__global__ void __launch_bounds__(256)
gemm16_bias(const uint16_t* __restrict__ A, const uint16_t* __restrict__ W,
            const float* __restrict__ bias, float* __restrict__ out) {
    __shared__ __align__(16) uint16_t lds_a[16 * 520];   // pitch 520 -> no bank conflicts
    const int tid = threadIdx.x;
    const int m0  = blockIdx.x * 16;
    for (int i = tid; i < 16 * 512 / 8; i += 256) {      // 16B chunks, coalesced
        const int r = i >> 6, kc = i & 63;
        const uint4 vdat = *(const uint4*)(A + (size_t)(m0 + r) * 512 + kc * 8);
        *(uint4*)(&lds_a[r * 520 + kc * 8]) = vdat;
    }
    __syncthreads();

    const int lane = tid & 31, wave = tid >> 5;
    const int mrow = lane & 15, kh = lane >> 4, nlo = lane & 15;
    v8f acc[4] = {};

    v16bf bfrag[4];
#pragma unroll
    for (int j = 0; j < 4; ++j) {
        const int col = (wave + j * 8) * 16 + nlo;
        bfrag[j] = load_frag(&W[(size_t)col * 512 + kh * 8]);
    }
#pragma unroll
    for (int kb = 0; kb < 16; ++kb) {
        const int k0 = kb * 32;
        const v16bf afrag = load_frag(&lds_a[mrow * 520 + k0 + kh * 8]);
        v16bf bnext[4];
        if (kb < 15) {
#pragma unroll
            for (int j = 0; j < 4; ++j) {
                const int col = (wave + j * 8) * 16 + nlo;
                bnext[j] = load_frag(&W[(size_t)col * 512 + k0 + 32 + kh * 8]);
            }
        }
        SCHED_FENCE();   // keep next-step loads above this point
#pragma unroll
        for (int j = 0; j < 4; ++j)
            acc[j] = __builtin_amdgcn_wmma_f32_16x16x32_bf16(
                false, afrag, false, bfrag[j], (short)0, acc[j], false, false);
        if (kb < 15) {
#pragma unroll
            for (int j = 0; j < 4; ++j) bfrag[j] = bnext[j];
        }
    }
#pragma unroll
    for (int j = 0; j < 4; ++j) {
        const int col = (wave + j * 8) * 16 + nlo;
        const float bs = bias[col];
#pragma unroll
        for (int v = 0; v < 8; ++v) {
            const int row = m0 + kh * 8 + v;
            out[(size_t)row * 512 + col] = acc[j][v] + bs;
        }
    }
}

// ---------------------------------------------------------------------------
// K4: fused av GEMM + (+ba +ah, tanh, *Wd, row-reduce) -> scores[M]
//     32-row M-tiles: each B-fragment feeds MT=2 WMMAs; B fragments are
//     software-pipelined one k-step ahead with a sched_barrier so WMMAs wait
//     behind a non-zero loadcnt instead of loadcnt==0.
//     grid = MM/32 = 392 blocks, block = 256 (8 waves).
// ---------------------------------------------------------------------------
__global__ void __launch_bounds__(256)
fused_av_score(const uint16_t* __restrict__ Abf, const uint16_t* __restrict__ Wa,
               const float* __restrict__ ba, const float* __restrict__ ah,
               const float* __restrict__ Wd, const float* __restrict__ bd,
               float* __restrict__ scores) {
    __shared__ __align__(16) uint16_t lds_a[MT * 16 * 520];
    __shared__ float lds_part[8][MT * 16];
    const int tid = threadIdx.x;
    const int m0  = blockIdx.x * (MT * 16);

    // stage A tile (32 rows x 512 K, bf16) in 16B chunks, coalesced
    for (int i = tid; i < MT * 16 * 512 / 8; i += 256) {
        const int r = i >> 6, kc = i & 63;
        const uint4 vdat = *(const uint4*)(Abf + (size_t)(m0 + r) * 512 + kc * 8);
        *(uint4*)(&lds_a[r * 520 + kc * 8]) = vdat;
    }
    __syncthreads();

    const int lane = tid & 31, wave = tid >> 5;
    const int mrow = lane & 15, kh = lane >> 4, nlo = lane & 15;
    v8f acc[4][MT] = {};

    v16bf bfrag[4];
#pragma unroll
    for (int j = 0; j < 4; ++j) {
        const int col = (wave + j * 8) * 16 + nlo;
        bfrag[j] = load_frag(&Wa[(size_t)col * 512 + kh * 8]);
    }
#pragma unroll
    for (int kb = 0; kb < 16; ++kb) {
        const int k0 = kb * 32;
        v16bf afrag[MT];
#pragma unroll
        for (int mt = 0; mt < MT; ++mt)
            afrag[mt] = load_frag(&lds_a[(mt * 16 + mrow) * 520 + k0 + kh * 8]);
        v16bf bnext[4];
        if (kb < 15) {
#pragma unroll
            for (int j = 0; j < 4; ++j) {
                const int col = (wave + j * 8) * 16 + nlo;
                bnext[j] = load_frag(&Wa[(size_t)col * 512 + k0 + 32 + kh * 8]);
            }
        }
        SCHED_FENCE();   // keep next-step loads above this point
#pragma unroll
        for (int j = 0; j < 4; ++j)
#pragma unroll
            for (int mt = 0; mt < MT; ++mt)
                acc[j][mt] = __builtin_amdgcn_wmma_f32_16x16x32_bf16(
                    false, afrag[mt], false, bfrag[j], (short)0, acc[j][mt], false, false);
        if (kb < 15) {
#pragma unroll
            for (int j = 0; j < 4; ++j) bfrag[j] = bnext[j];
        }
    }

    float part[MT][8] = {};
#pragma unroll
    for (int j = 0; j < 4; ++j) {
        const int col  = (wave + j * 8) * 16 + nlo;
        const float bav = ba[col];
        const float wd  = Wd[col];
#pragma unroll
        for (int mt = 0; mt < MT; ++mt) {
#pragma unroll
            for (int v = 0; v < 8; ++v) {
                const int m = m0 + mt * 16 + kh * 8 + v;
                const int b = m / NN;
                const float val = acc[j][mt][v] + bav + ah[(size_t)b * 512 + col];
                part[mt][v] += tanhf(val) * wd;
            }
        }
    }
    // reduce over the 16 lanes of each half-wave (columns of the tile)
#pragma unroll
    for (int mt = 0; mt < MT; ++mt) {
#pragma unroll
        for (int v = 0; v < 8; ++v) {
            float s = part[mt][v];
            s += __shfl_xor(s, 1, 32);
            s += __shfl_xor(s, 2, 32);
            s += __shfl_xor(s, 4, 32);
            s += __shfl_xor(s, 8, 32);
            part[mt][v] = s;
        }
    }
    if (nlo == 0) {
#pragma unroll
        for (int mt = 0; mt < MT; ++mt)
#pragma unroll
            for (int v = 0; v < 8; ++v)
                lds_part[wave][mt * 16 + kh * 8 + v] = part[mt][v];
    }
    __syncthreads();
    if (tid < MT * 16) {
        float s = bd[0];
#pragma unroll
        for (int w = 0; w < 8; ++w) s += lds_part[w][tid];
        scores[m0 + tid] = s;
    }
}

// ---------------------------------------------------------------------------
// K5: softmax over 49 positions per batch.  grid = B, block = 64.
// ---------------------------------------------------------------------------
__global__ void __launch_bounds__(64)
softmax_kernel(const float* __restrict__ scores, float* __restrict__ weights) {
    __shared__ float tmp[64];
    const int b = blockIdx.x, t = threadIdx.x;
    const float v = (t < NN) ? scores[b * NN + t] : -3.4e38f;
    tmp[t] = v;
    __syncthreads();
    for (int s = 32; s > 0; s >>= 1) { if (t < s) tmp[t] = fmaxf(tmp[t], tmp[t + s]); __syncthreads(); }
    const float mx = tmp[0];
    __syncthreads();
    const float e = (t < NN) ? expf(v - mx) : 0.f;
    tmp[t] = e;
    __syncthreads();
    for (int s = 32; s > 0; s >>= 1) { if (t < s) tmp[t] += tmp[t + s]; __syncthreads(); }
    if (t < NN) weights[b * NN + t] = e / tmp[0];
}

// ---------------------------------------------------------------------------
// K6: att_res[b,c] = sum_n a[b,n,c] * w[b,n].  grid = B, block = 256.
// ---------------------------------------------------------------------------
__global__ void __launch_bounds__(256)
attres_kernel(const float* __restrict__ a_f32, const float* __restrict__ weights,
              float* __restrict__ out) {
    __shared__ float w[NN];
    const int b = blockIdx.x, t = threadIdx.x;
    if (t < NN) w[t] = weights[b * NN + t];
    __syncthreads();
    for (int c = t; c < CC; c += 256) {
        const float* base = a_f32 + (size_t)b * NN * CC + c;
        float acc = 0.f;
#pragma unroll 7
        for (int n = 0; n < NN; ++n) acc += base[(size_t)n * CC] * w[n];
        out[(size_t)b * CC + c] = acc;
    }
}

// ---------------------------------------------------------------------------
extern "C" void kernel_launch(void* const* d_in, const int* in_sizes, int n_in,
                              void* d_out, int out_size, void* d_ws, size_t ws_size,
                              hipStream_t stream) {
    const float* att_v = (const float*)d_in[0];
    const float* h     = (const float*)d_in[1];
    const float* Wa    = (const float*)d_in[2];
    const float* ba    = (const float*)d_in[3];
    const float* Wh    = (const float*)d_in[4];
    const float* bh    = (const float*)d_in[5];
    const float* Wd    = (const float*)d_in[6];
    const float* bd    = (const float*)d_in[7];
    float* out = (float*)d_out;

    char* ws = (char*)d_ws;
    float*    a_f32   = (float*)   (ws + 0);              // 12544*512*4 = 25,690,112
    uint16_t* a_bf16  = (uint16_t*)(ws + 25690112);       // 12,845,056
    uint16_t* wa_b    = (uint16_t*)(ws + 38535168);       //    524,288
    uint16_t* wh_b    = (uint16_t*)(ws + 39059456);       //    524,288
    uint16_t* h_b     = (uint16_t*)(ws + 39583744);       //    262,144
    float*    ah      = (float*)   (ws + 39845888);       //    524,288
    float*    scores  = (float*)   (ws + 40370176);       //     50,176
    float*    weights = (float*)   (ws + 40420352);       //     50,176  (total ~40.5 MB)

    // 1) pool -> a (f32 + bf16)
    pool_kernel<<<BB * 8, 256, 0, stream>>>(att_v, a_f32, a_bf16);

    // 2) convert weights / h to bf16
    cvt_kernel<<<(512 * 512 + 255) / 256, 256, 0, stream>>>(Wa, wa_b, 512 * 512);
    cvt_kernel<<<(512 * 512 + 255) / 256, 256, 0, stream>>>(Wh, wh_b, 512 * 512);
    cvt_kernel<<<(BB * 512 + 255) / 256, 256, 0, stream>>>(h,  h_b,  BB * 512);

    // 3) ah = h @ Wh^T + bh   (M = 256)
    gemm16_bias<<<BB / 16, 256, 0, stream>>>(h_b, wh_b, bh, ah);

    // 4) fused av GEMM + tanh/Wd epilogue -> scores  (M = 12544, 32-row tiles)
    fused_av_score<<<MM / (MT * 16), 256, 0, stream>>>(a_bf16, wa_b, ba, ah, Wd, bd, scores);

    // 5) softmax per batch
    softmax_kernel<<<BB, 64, 0, stream>>>(scores, weights);

    // 6) att_res = a^T @ weight
    attres_kernel<<<BB, 256, 0, stream>>>(a_f32, weights, out);
}
